// Attention_70660801954478
// MI455X (gfx1250) — matrix operand order repaired and checked by reading, not compiled, not run on hardware
//
#include <hip/hip_runtime.h>
#include <math.h>

#define HQ   32
#define KVH  8
#define DHD  64
#define WIN  256

// per-wave LDS row strides (ushorts): 16B-aligned, bank-conflict-free b128 reads
#define QROW 72
#define PROW 40

#define LOG2E    1.4426950408889634f
#define NEG_BIG  (-1.0e30f)
#define EXP_SH   12.0f                  // fixed softmax shift (cancels in normalization)
// log2(10000)/32
#define ROPE_L2  0.41524101186092027f

typedef __attribute__((ext_vector_type(16))) __bf16 v16bf;
typedef __attribute__((ext_vector_type(2)))  __bf16 v2bf;
typedef __attribute__((ext_vector_type(8)))  float  v8f;

union FragA { v16bf bf; uint4 u[2]; };

__device__ __forceinline__ unsigned pack2(float lo, float hi) {
  v2bf t = {(__bf16)lo, (__bf16)hi};
  return __builtin_bit_cast(unsigned, t);
}

#define WMMA_BF16(A, B, C) \
  __builtin_amdgcn_wmma_f32_16x16x32_bf16(false, (A), false, (B), (short)0, (C), false, false)

// ---------------- pre-pass 1: RoPE + bf16 K -> wsK [b][kvh][s][64] ----------------
__global__ __launch_bounds__(128)
void rope_k_kernel(const float* __restrict__ K, unsigned short* __restrict__ wsK, int S) {
  const int tid = threadIdx.x;
  const int R   = blockIdx.x * 32 + (tid >> 2);   // global (b,kvh,s) row
  const int d0  = (tid & 3) * 8;                  // pair-dim slice
  const int s   = R % S;
  const int bh  = R / S;                          // b*KVH + kvh
  const int bq  = bh / KVH, kvh = bh % KVH;
  const float* krow = K + ((size_t)bq * S + s) * (KVH * DHD) + kvh * DHD;
  unsigned short* orow = wsK + (size_t)R * DHD;
  const float pos = (float)s;
#pragma unroll
  for (int jj = 0; jj < 8; jj += 2) {
    float o1[2], o2[2];
#pragma unroll
    for (int e = 0; e < 2; ++e) {
      int d = d0 + jj + e;
      float invf = exp2f(-(float)d * ROPE_L2);
      float sn, cs;
      __sincosf(pos * invf, &sn, &cs);
      float x1 = krow[d], x2 = krow[d + 32];
      o1[e] = x1 * cs - x2 * sn;
      o2[e] = x2 * cs + x1 * sn;
    }
    *(unsigned*)&orow[d0 + jj]      = pack2(o1[0], o1[1]);
    *(unsigned*)&orow[d0 + jj + 32] = pack2(o2[0], o2[1]);
  }
}

// ---- pre-pass 2: bf16 V, transposed + key-interleaved: wsV [b][kvh][t][d][32] ----
// column(r) = 2*(r&15) + (r>=16)  -> matches P-store interleave in the main kernel
__global__ __launch_bounds__(128)
void conv_v_kernel(const float* __restrict__ V, unsigned short* __restrict__ wsV, int S) {
  const int tid = threadIdx.x;
  const int i   = tid >> 3;          // 0..15
  const int vd0 = (tid & 7) * 8;
  const int t   = blockIdx.x;
  const int kvh = blockIdx.y;
  const int bq  = blockIdx.z;
  const int kt  = t * 32;
  const float* vrow0 = V + ((size_t)bq * S + kt + i) * (KVH * DHD) + kvh * DHD;
  const float* vrow1 = vrow0 + 16 * (KVH * DHD);
  unsigned short* otile = wsV + ((size_t)(bq * KVH + kvh) * (S / 32) + t) * (32 * DHD);
#pragma unroll
  for (int jj = 0; jj < 8; ++jj) {
    *(unsigned*)&otile[(vd0 + jj) * 32 + 2 * i] = pack2(vrow0[vd0 + jj], vrow1[vd0 + jj]);
  }
}

// --------------------------- main attention kernel ---------------------------
__global__ __launch_bounds__(256)
void fa_win_kernel(const float* __restrict__ Q,
                   const unsigned short* __restrict__ wsK,
                   const unsigned short* __restrict__ wsV,
                   float* __restrict__ O, int S) {
  __shared__ unsigned short qbuf[8][16 * QROW];
  __shared__ unsigned short pbuf[8][16 * PROW];

  const int tid  = threadIdx.x;
  const int lane = tid & 31;
  const int wvid = tid >> 5;
  const int hi16 = lane >> 4;
  const int l15  = lane & 15;

  const int q0  = blockIdx.x * 128;
  const int h   = blockIdx.y;
  const int b   = blockIdx.z;
  const int kvh = h >> 2;
  const int qw  = q0 + wvid * 16;      // this wave's 16-query tile (independent wave)

  const float* Qbase = Q + (size_t)b * S * (HQ * DHD) + h * DHD;
  const unsigned short* kbh = wsK + (size_t)(b * KVH + kvh) * S * DHD;
  const unsigned short* vbh = wsV + (size_t)(b * KVH + kvh) * (S / 32) * (32 * DHD);

  // ---- Q tile: RoPE + 1/sqrt(D) -> per-wave LDS -> A fragments ----
  {
    const int r  = lane >> 1;
    const int d0 = (lane & 1) * 16;
    const float* qrow = Qbase + (size_t)(qw + r) * (HQ * DHD);
    const float pos = (float)(qw + r);
#pragma unroll
    for (int jj = 0; jj < 16; jj += 2) {
      float o1[2], o2[2];
#pragma unroll
      for (int e = 0; e < 2; ++e) {
        int d = d0 + jj + e;
        float invf = exp2f(-(float)d * ROPE_L2);
        float sn, cs;
        __sincosf(pos * invf, &sn, &cs);
        float x1 = qrow[d], x2 = qrow[d + 32];
        o1[e] = (x1 * cs - x2 * sn) * 0.125f;
        o2[e] = (x2 * cs + x1 * sn) * 0.125f;
      }
      *(unsigned*)&qbuf[wvid][r * QROW + d0 + jj]      = pack2(o1[0], o1[1]);
      *(unsigned*)&qbuf[wvid][r * QROW + d0 + jj + 32] = pack2(o2[0], o2[1]);
    }
  }
  asm volatile("s_wait_dscnt 0" ::: "memory");

  FragA aq0, aq1;
  {
    const unsigned short* base = &qbuf[wvid][l15 * QROW];
    aq0.u[0] = *(const uint4*)(base +      16 * hi16);
    aq0.u[1] = *(const uint4*)(base + 8  + 16 * hi16);
    aq1.u[0] = *(const uint4*)(base + 32 + 16 * hi16);
    aq1.u[1] = *(const uint4*)(base + 40 + 16 * hi16);
  }

  FragA onesf;
  {
    uint4 o = {0x3F803F80u, 0x3F803F80u, 0x3F803F80u, 0x3F803F80u};
    onesf.u[0] = o; onesf.u[1] = o;
  }

  v8f acc[4], accl;
  {
    v8f z = {};
#pragma unroll
    for (int g = 0; g < 4; ++g) acc[g] = z;
    accl = z;
  }

  // per-wave exact window: keys [max(0,qw-WIN), qw+15], tiles 32-aligned
  int kstart = qw - WIN; if (kstart < 0) kstart = 0;
  kstart &= ~31;

  for (int kt = kstart; kt <= qw + 15; kt += 32) {
    // ---- K B-fragments straight from global (L2-resident bf16) ----
    const unsigned short* kb0 = kbh + (size_t)(kt + l15) * DHD + 16 * hi16;
    const unsigned short* kb1 = kb0 + 16 * DHD;
    FragA b00, b01, b10, b11;
    b00.u[0] = *(const uint4*)(kb0);      b00.u[1] = *(const uint4*)(kb0 + 8);
    b01.u[0] = *(const uint4*)(kb0 + 32); b01.u[1] = *(const uint4*)(kb0 + 40);
    b10.u[0] = *(const uint4*)(kb1);      b10.u[1] = *(const uint4*)(kb1 + 8);
    b11.u[0] = *(const uint4*)(kb1 + 32); b11.u[1] = *(const uint4*)(kb1 + 40);

    v8f sc0, sc1;
    {
      v8f z = {};
      sc0 = WMMA_BF16(aq0.bf, b00.bf, z);
      sc1 = WMMA_BF16(aq0.bf, b10.bf, z);
      sc0 = WMMA_BF16(aq1.bf, b01.bf, sc0);
      sc1 = WMMA_BF16(aq1.bf, b11.bf, sc1);
    }

    // ---- fixed-shift softmax numerators; mask only on boundary tiles ----
    unsigned short* pb = &pbuf[wvid][(8 * hi16) * PROW + 2 * l15];
    if (!((kt + 31 <= qw) && (kt >= qw + 15 - WIN))) {
      const int dbase = qw + 8 * hi16 - kt - l15;   // (i - k0) - v
#pragma unroll
      for (int v = 0; v < 8; ++v) {
        float a0 = __builtin_fmaf(sc0[v], LOG2E, -EXP_SH);
        float a1 = __builtin_fmaf(sc1[v], LOG2E, -EXP_SH);
        a0 = ((unsigned)(dbase + v)      <= (unsigned)WIN) ? a0 : NEG_BIG;
        a1 = ((unsigned)(dbase + v - 16) <= (unsigned)WIN) ? a1 : NEG_BIG;
        *(unsigned*)(pb + v * PROW) = pack2(exp2f(a0), exp2f(a1));
      }
    } else {
#pragma unroll
      for (int v = 0; v < 8; ++v) {
        float p0 = exp2f(__builtin_fmaf(sc0[v], LOG2E, -EXP_SH));
        float p1 = exp2f(__builtin_fmaf(sc1[v], LOG2E, -EXP_SH));
        *(unsigned*)(pb + v * PROW) = pack2(p0, p1);
      }
    }
    asm volatile("s_wait_dscnt 0" ::: "memory");

    FragA ap;
    {
      const unsigned short* pr = &pbuf[wvid][l15 * PROW];
      ap.u[0] = *(const uint4*)(pr +      16 * hi16);
      ap.u[1] = *(const uint4*)(pr + 8  + 16 * hi16);
    }
    // row sums first (depends only on ap), V fragments load underneath
    accl = WMMA_BF16(ap.bf, onesf.bf, accl);

    // ---- V B-fragments straight from global (pre-transposed, interleaved) ----
    const unsigned short* vt = vbh + (size_t)(kt >> 5) * (32 * DHD) + l15 * 32 + 16 * hi16;
    FragA bv0, bv1, bv2, bv3;
    bv0.u[0] = *(const uint4*)(vt);         bv0.u[1] = *(const uint4*)(vt + 8);
    bv1.u[0] = *(const uint4*)(vt + 512);   bv1.u[1] = *(const uint4*)(vt + 520);
    bv2.u[0] = *(const uint4*)(vt + 1024);  bv2.u[1] = *(const uint4*)(vt + 1032);
    bv3.u[0] = *(const uint4*)(vt + 1536);  bv3.u[1] = *(const uint4*)(vt + 1544);

    acc[0] = WMMA_BF16(ap.bf, bv0.bf, acc[0]);
    acc[1] = WMMA_BF16(ap.bf, bv1.bf, acc[1]);
    acc[2] = WMMA_BF16(ap.bf, bv2.bf, acc[2]);
    acc[3] = WMMA_BF16(ap.bf, bv3.bf, acc[3]);
  }

  // ---- normalize and store ----
#pragma unroll
  for (int v = 0; v < 8; ++v) {
    const int m = v + 8 * hi16;
    const float inv = __builtin_amdgcn_rcpf(accl[v]);
    float* orow = O + ((size_t)b * S + (qw + m)) * (HQ * DHD) + h * DHD;
#pragma unroll
    for (int g = 0; g < 4; ++g) {
      orow[16 * g + l15] = acc[g][v] * inv;
    }
  }
}

extern "C" void kernel_launch(void* const* d_in, const int* in_sizes, int n_in,
                              void* d_out, int out_size, void* d_ws, size_t ws_size,
                              hipStream_t stream) {
  const float* Q = (const float*)d_in[0];
  const float* K = (const float*)d_in[1];
  const float* V = (const float*)d_in[2];
  float* O = (float*)d_out;

  const int B = 2;
  const int S = in_sizes[0] / (B * HQ * DHD);   // 2048

  // workspace: bf16 K (RoPE'd) + bf16 V (transposed) = 2 * B*KVH*S*64*2 bytes = 8 MB
  unsigned short* wsK = (unsigned short*)d_ws;
  unsigned short* wsV = wsK + (size_t)B * KVH * S * DHD;

  rope_k_kernel<<<dim3((B * KVH * S) / 32), dim3(128), 0, stream>>>(K, wsK, S);
  conv_v_kernel<<<dim3(S / 32, KVH, B), dim3(128), 0, stream>>>(V, wsV, S);
  fa_win_kernel<<<dim3(S / 128, HQ, B), dim3(256), 0, stream>>>(Q, wsK, wsV, O, S);
}
